// Multi_Head_Attention_71786083385456
// MI455X (gfx1250) — matrix operand order repaired and checked by reading, not compiled, not run on hardware
//
#include <hip/hip_runtime.h>

// ---------------------------------------------------------------------------
// MHA layer for MI455X (gfx1250): bf16 WMMA, flash attention, no-LDS GEMMs.
// Shapes: B=4, L=2048, D=512, H=8, dh=64.  Faithful to torch's flat
// view(B*H, L, dh): group g is a contiguous [2048,64] slab of the flat buffer.
// ---------------------------------------------------------------------------

#define BATCH   4
#define SEQ     2048
#define DMODEL  512
#define NHEAD   8
#define DH      64
#define MROWS   (BATCH * SEQ)          // 8192
#define NGROUP  (BATCH * NHEAD)        // 32
#define GSTRIDE (SEQ * DH)             // 131072 elements per group slab
#define WELEMS  (DMODEL * DMODEL)      // 262144 elements per weight matrix

typedef __bf16 bf16x16 __attribute__((ext_vector_type(16)));
typedef float  f32x8   __attribute__((ext_vector_type(8)));

__device__ __forceinline__ unsigned short f2bf(float f) {
  union { float f; unsigned u; } c; c.f = f;
  unsigned u = c.u;
  return (unsigned short)((u + 0x7FFFu + ((u >> 16) & 1u)) >> 16);
}

__device__ __forceinline__ f32x8 wmma_bf16(bf16x16 a, bf16x16 b, f32x8 c) {
  return __builtin_amdgcn_wmma_f32_16x16x32_bf16(
      /*neg_a=*/false, a, /*neg_b=*/false, b,
      /*c_mod=*/(short)0, c, /*reuse_a=*/false, /*reuse_b=*/false);
}

// ---- fragment loaders -------------------------------------------------------
// A 16x32 (MxK): lane = (M&15) + 16*kh; elems 0..15 <- K = 16*kh + e.
//                Two 16B chunks.
__device__ __forceinline__ bf16x16 gA_bf(const unsigned short* A, int lda,
                                         int m0, int k0, int lane) {
  int m = m0 + (lane & 15), kh = (lane >> 4) * 16;
  const unsigned short* p = A + (size_t)m * lda + k0 + kh;
  union { bf16x16 v; uint4 q[2]; } u;
  u.q[0] = *(const uint4*)(p);
  u.q[1] = *(const uint4*)(p + 8);
  return u.v;
}

__device__ __forceinline__ bf16x16 ldsA_bf(const unsigned short* A, int lda,
                                           int k0, int lane) {
  int m = lane & 15, kh = (lane >> 4) * 16;
  const unsigned short* p = A + m * lda + k0 + kh;
  union { bf16x16 v; uint4 q[2]; } u;
  u.q[0] = *(const uint4*)(p);
  u.q[1] = *(const uint4*)(p + 8);
  return u.v;
}

// B 32x16 (KxN): lane = (N&15) + 16*kg; elems 0..15 <- K = 16*kg .. +15.
// Source rows are N with K contiguous -> one 32B chunk per lane.
__device__ __forceinline__ bf16x16 gB_bf(const unsigned short* Bm, int ldb,
                                         int n0, int k0, int lane) {
  int n = n0 + (lane & 15), kb = (lane >> 4) * 16;
  const unsigned short* p = Bm + (size_t)n * ldb + k0 + kb;
  union { bf16x16 v; uint4 q[2]; } u;
  u.q[0] = *(const uint4*)(p);
  u.q[1] = *(const uint4*)(p + 8);
  return u.v;
}

__device__ __forceinline__ bf16x16 ldsB_bf(const unsigned short* Bm, int ldb,
                                           int k0, int lane) {
  int n = lane & 15, kb = (lane >> 4) * 16;
  const unsigned short* p = Bm + n * ldb + k0 + kb;
  union { bf16x16 v; uint4 q[2]; } u;
  u.q[0] = *(const uint4*)(p);
  u.q[1] = *(const uint4*)(p + 8);
  return u.v;
}

// ---------------------------------------------------------------------------
// Kernel 0: one-shot f32 -> bf16 conversion (bandwidth bound, ~1us total).
// ---------------------------------------------------------------------------
__global__ __launch_bounds__(256) void k_cvt(const float* __restrict__ in,
                                             unsigned short* __restrict__ out) {
  size_t i = ((size_t)blockIdx.x * 256 + threadIdx.x) * 8;
  float4 f0 = *(const float4*)(in + i);
  float4 f1 = *(const float4*)(in + i + 4);
  union { uint4 q; unsigned short s[8]; } u;
  u.s[0] = f2bf(f0.x); u.s[1] = f2bf(f0.y); u.s[2] = f2bf(f0.z); u.s[3] = f2bf(f0.w);
  u.s[4] = f2bf(f1.x); u.s[5] = f2bf(f1.y); u.s[6] = f2bf(f1.z); u.s[7] = f2bf(f1.w);
  *(uint4*)(out + i) = u.q;
}

// ---------------------------------------------------------------------------
// Kernel 1: QKV projections.  C = x @ W^T + b, all-bf16 operands, no LDS:
// W is 0.5MB bf16 (L2-resident), fragments are contiguous 32B global chunks.
// BM=128 (8 waves x 16 rows), BN=128 (8 N-tiles/wave), BK=32.
// ---------------------------------------------------------------------------
__global__ __launch_bounds__(256) void k_qkv(
    const unsigned short* __restrict__ xb,
    const unsigned short* __restrict__ Wqb, const unsigned short* __restrict__ Wkb,
    const unsigned short* __restrict__ Wvb,
    const float* __restrict__ bq, const float* __restrict__ bk,
    const float* __restrict__ bv,
    unsigned short* __restrict__ Qb, unsigned short* __restrict__ Kb,
    unsigned short* __restrict__ Vb) {
  const unsigned short* W; const float* bias; unsigned short* out;
  if (blockIdx.z == 0)      { W = Wqb; bias = bq; out = Qb; }
  else if (blockIdx.z == 1) { W = Wkb; bias = bk; out = Kb; }
  else                      { W = Wvb; bias = bv; out = Vb; }

  const int lane = threadIdx.x & 31, wave = threadIdx.x >> 5;
  const int m0 = blockIdx.y * 128 + wave * 16;
  const int n0 = blockIdx.x * 128;

  const f32x8 zero = {0, 0, 0, 0, 0, 0, 0, 0};
  f32x8 acc[8];
#pragma unroll
  for (int i = 0; i < 8; ++i) acc[i] = zero;

#pragma unroll 2
  for (int k0 = 0; k0 < DMODEL; k0 += 32) {
    bf16x16 a = gA_bf(xb, DMODEL, m0, k0, lane);
    // independent fragment registers -> loads overlap with WMMAs
    bf16x16 b4[4];
#pragma unroll
    for (int nt = 0; nt < 4; ++nt) b4[nt] = gB_bf(W, DMODEL, n0 + nt * 16, k0, lane);
#pragma unroll
    for (int nt = 0; nt < 4; ++nt) acc[nt] = wmma_bf16(a, b4[nt], acc[nt]);
#pragma unroll
    for (int nt = 0; nt < 4; ++nt) b4[nt] = gB_bf(W, DMODEL, n0 + 64 + nt * 16, k0, lane);
#pragma unroll
    for (int nt = 0; nt < 4; ++nt) acc[4 + nt] = wmma_bf16(a, b4[nt], acc[4 + nt]);
  }

  const int rb = 8 * (lane >> 4);
#pragma unroll
  for (int nt = 0; nt < 8; ++nt) {
    int col = n0 + nt * 16 + (lane & 15);
    float b = bias[col];
#pragma unroll
    for (int r = 0; r < 8; ++r)
      out[(size_t)(m0 + rb + r) * DMODEL + col] = f2bf(acc[nt][r] + b);
  }
}

// ---------------------------------------------------------------------------
// Kernel 2: flash attention per flat group g (contiguous [2048,64] slab).
// 128 query rows per WG (wave w owns 16), 64-key tiles, online softmax.
// ---------------------------------------------------------------------------
__global__ __launch_bounds__(256) void k_attn(
    const unsigned short* __restrict__ Qb, const unsigned short* __restrict__ Kb,
    const unsigned short* __restrict__ Vb, unsigned short* __restrict__ Cb) {
  const int lane = threadIdx.x & 31, wave = threadIdx.x >> 5;
  const int g = blockIdx.y;
  const int m0 = blockIdx.x * 128 + wave * 16;
  const unsigned short* Qg = Qb + (size_t)g * GSTRIDE;
  const unsigned short* Kg = Kb + (size_t)g * GSTRIDE;
  const unsigned short* Vg = Vb + (size_t)g * GSTRIDE;
  unsigned short*       Cg = Cb + (size_t)g * GSTRIDE;

  __shared__ __align__(16) float          Sws[8][16][64];   // 32 KB scores
  __shared__ __align__(16) unsigned short Pws[128][64];     // 16 KB probs bf16
  __shared__ __align__(16) unsigned short Vt[64][64];       //  8 KB V^T tile
  __shared__ float rowM[128], rowL[128], rowC[128];

  // Q fragments resident in VGPRs for the whole kernel (dh=64 -> 2 k-steps).
  bf16x16 aq0 = gA_bf(Qg, DH, m0, 0, lane);
  bf16x16 aq1 = gA_bf(Qg, DH, m0, 32, lane);

  const f32x8 zero = {0, 0, 0, 0, 0, 0, 0, 0};
  f32x8 o[4];
#pragma unroll
  for (int i = 0; i < 4; ++i) o[i] = zero;

  if (threadIdx.x < 128) { rowM[threadIdx.x] = -3.0e38f; rowL[threadIdx.x] = 0.f; }

  const int rb = 8 * (lane >> 4);

  for (int kb0 = 0; kb0 < SEQ; kb0 += 64) {
    // prefetch next K/V tile rows into cache (global_prefetch path)
    if (kb0 + 64 < SEQ && threadIdx.x < 64) {
      __builtin_prefetch(Kg + (size_t)(kb0 + 64 + threadIdx.x) * DH, 0, 1);
      __builtin_prefetch(Vg + (size_t)(kb0 + 64 + threadIdx.x) * DH, 0, 1);
    }
    __syncthreads();
    { // stage V^T tile: Vt[d][key] = V[kb0+key][d]
      int key = threadIdx.x & 63, dbase = (threadIdx.x >> 6) * 16;
      const unsigned short* vp = Vg + (size_t)(kb0 + key) * DH + dbase;
      union { uint4 q; unsigned short s[8]; } u0, u1;
      u0.q = *(const uint4*)(vp);
      u1.q = *(const uint4*)(vp + 8);
#pragma unroll
      for (int i = 0; i < 8; ++i) {
        Vt[dbase + i][key]     = u0.s[i];
        Vt[dbase + 8 + i][key] = u1.s[i];
      }
    }
    // S = (Q K^T) * scale; K fragments straight from global (K-major rows).
    {
      f32x8 s[4];
      bf16x16 kf[4];
#pragma unroll
      for (int nt = 0; nt < 4; ++nt) kf[nt] = gB_bf(Kg, DH, kb0 + nt * 16, 0, lane);
#pragma unroll
      for (int nt = 0; nt < 4; ++nt) s[nt] = wmma_bf16(aq0, kf[nt], zero);
#pragma unroll
      for (int nt = 0; nt < 4; ++nt) kf[nt] = gB_bf(Kg, DH, kb0 + nt * 16, 32, lane);
#pragma unroll
      for (int nt = 0; nt < 4; ++nt) s[nt] = wmma_bf16(aq1, kf[nt], s[nt]);
#pragma unroll
      for (int nt = 0; nt < 4; ++nt) {
        int col = nt * 16 + (lane & 15);
#pragma unroll
        for (int r = 0; r < 8; ++r)
          Sws[wave][rb + r][col] = s[nt][r] * 0.125f;   // dh^-0.5 = 1/8
      }
    }
    __syncthreads();
    // online softmax: one thread per query row
    if (threadIdx.x < 128) {
      const float* sp = &Sws[threadIdx.x >> 4][threadIdx.x & 15][0];
      float mold = rowM[threadIdx.x];
      float mnew = mold;
#pragma unroll 8
      for (int c = 0; c < 64; ++c) mnew = fmaxf(mnew, sp[c]);
      float corr = __expf(mold - mnew);
      float l = rowL[threadIdx.x] * corr;
#pragma unroll 8
      for (int c = 0; c < 64; ++c) {
        float p = __expf(sp[c] - mnew);
        l += p;
        Pws[threadIdx.x][c] = f2bf(p);
      }
      rowM[threadIdx.x] = mnew; rowL[threadIdx.x] = l; rowC[threadIdx.x] = corr;
    }
    __syncthreads();
    { // rescale running O, then O += P @ V
      float cf[8];
#pragma unroll
      for (int r = 0; r < 8; ++r) cf[r] = rowC[wave * 16 + rb + r];
#pragma unroll
      for (int nt = 0; nt < 4; ++nt)
#pragma unroll
        for (int r = 0; r < 8; ++r) o[nt][r] *= cf[r];

      bf16x16 ap0 = ldsA_bf(&Pws[wave * 16][0], 64, 0, lane);
      bf16x16 ap1 = ldsA_bf(&Pws[wave * 16][0], 64, 32, lane);
      bf16x16 vf[4];
#pragma unroll
      for (int nt = 0; nt < 4; ++nt) vf[nt] = ldsB_bf(&Vt[nt * 16][0], 64, 0, lane);
#pragma unroll
      for (int nt = 0; nt < 4; ++nt) o[nt] = wmma_bf16(ap0, vf[nt], o[nt]);
#pragma unroll
      for (int nt = 0; nt < 4; ++nt) vf[nt] = ldsB_bf(&Vt[nt * 16][0], 64, 32, lane);
#pragma unroll
      for (int nt = 0; nt < 4; ++nt) o[nt] = wmma_bf16(ap1, vf[nt], o[nt]);
    }
  }
  __syncthreads();
  { // normalize by running sum, emit bf16 context
    float il[8];
#pragma unroll
    for (int r = 0; r < 8; ++r) il[r] = 1.0f / rowL[wave * 16 + rb + r];
#pragma unroll
    for (int nt = 0; nt < 4; ++nt) {
      int col = nt * 16 + (lane & 15);
#pragma unroll
      for (int r = 0; r < 8; ++r)
        Cg[(size_t)(m0 + rb + r) * DH + col] = f2bf(o[nt][r] * il[r]);
    }
  }
}

// ---------------------------------------------------------------------------
// Kernel 3: out = ctx @ Wo^T + bo + x   (f32 pre-LN buffer), no-LDS GEMM.
// ---------------------------------------------------------------------------
__global__ __launch_bounds__(256) void k_proj(
    const unsigned short* __restrict__ ctx, const unsigned short* __restrict__ Wob,
    const float* __restrict__ bo, const float* __restrict__ x,
    float* __restrict__ outp) {
  const int lane = threadIdx.x & 31, wave = threadIdx.x >> 5;
  const int m0 = blockIdx.y * 128 + wave * 16;
  const int n0 = blockIdx.x * 128;

  const f32x8 zero = {0, 0, 0, 0, 0, 0, 0, 0};
  f32x8 acc[8];
#pragma unroll
  for (int i = 0; i < 8; ++i) acc[i] = zero;

#pragma unroll 2
  for (int k0 = 0; k0 < DMODEL; k0 += 32) {
    bf16x16 a = gA_bf(ctx, DMODEL, m0, k0, lane);
    bf16x16 b4[4];
#pragma unroll
    for (int nt = 0; nt < 4; ++nt) b4[nt] = gB_bf(Wob, DMODEL, n0 + nt * 16, k0, lane);
#pragma unroll
    for (int nt = 0; nt < 4; ++nt) acc[nt] = wmma_bf16(a, b4[nt], acc[nt]);
#pragma unroll
    for (int nt = 0; nt < 4; ++nt) b4[nt] = gB_bf(Wob, DMODEL, n0 + 64 + nt * 16, k0, lane);
#pragma unroll
    for (int nt = 0; nt < 4; ++nt) acc[4 + nt] = wmma_bf16(a, b4[nt], acc[4 + nt]);
  }

  const int rb = 8 * (lane >> 4);
#pragma unroll
  for (int nt = 0; nt < 8; ++nt) {
    int col = n0 + nt * 16 + (lane & 15);
    float b = bo[col];
#pragma unroll
    for (int r = 0; r < 8; ++r) {
      size_t idx = (size_t)(m0 + rb + r) * DMODEL + col;
      outp[idx] = acc[nt][r] + b + x[idx];   // bias + residual
    }
  }
}

// ---------------------------------------------------------------------------
// Kernel 4: LayerNorm over D=512, one wave32 per row, shuffle reductions.
// ---------------------------------------------------------------------------
__global__ __launch_bounds__(256) void k_ln(
    const float* __restrict__ in, const float* __restrict__ gamma,
    const float* __restrict__ beta, float* __restrict__ out) {
  const int lane = threadIdx.x & 31, wave = threadIdx.x >> 5;
  const int row = blockIdx.x * 8 + wave;
  const float* p = in + (size_t)row * DMODEL + lane * 16;

  float v[16];
#pragma unroll
  for (int i = 0; i < 4; ++i) {
    float4 f = *(const float4*)(p + i * 4);
    v[i * 4 + 0] = f.x; v[i * 4 + 1] = f.y; v[i * 4 + 2] = f.z; v[i * 4 + 3] = f.w;
  }
  float s = 0.f;
#pragma unroll
  for (int i = 0; i < 16; ++i) s += v[i];
#pragma unroll
  for (int off = 16; off > 0; off >>= 1) s += __shfl_xor(s, off, 32);
  float mean = s * (1.0f / DMODEL);

  float q = 0.f;
#pragma unroll
  for (int i = 0; i < 16; ++i) { float d = v[i] - mean; q += d * d; }
#pragma unroll
  for (int off = 16; off > 0; off >>= 1) q += __shfl_xor(q, off, 32);
  float rstd = rsqrtf(q * (1.0f / DMODEL) + 1e-5f);

#pragma unroll
  for (int i = 0; i < 16; ++i) {
    int c = lane * 16 + i;
    out[(size_t)row * DMODEL + c] = (v[i] - mean) * rstd * gamma[c] + beta[c];
  }
}

// ---------------------------------------------------------------------------
extern "C" void kernel_launch(void* const* d_in, const int* in_sizes, int n_in,
                              void* d_out, int out_size, void* d_ws, size_t ws_size,
                              hipStream_t stream) {
  const float* x     = (const float*)d_in[0];
  const float* Wq    = (const float*)d_in[1];
  const float* bq    = (const float*)d_in[2];
  const float* Wk    = (const float*)d_in[3];
  const float* bk    = (const float*)d_in[4];
  const float* Wv    = (const float*)d_in[5];
  const float* bv    = (const float*)d_in[6];
  const float* Wo    = (const float*)d_in[7];
  const float* bo    = (const float*)d_in[8];
  const float* gamma = (const float*)d_in[9];
  const float* beta  = (const float*)d_in[10];
  float* out = (float*)d_out;

  const size_t NE = (size_t)MROWS * DMODEL;            // 8192*512
  char* w = (char*)d_ws;
  unsigned short* xb  = (unsigned short*)(w);          //  8 MB
  unsigned short* Qb  = xb + NE;                       //  8 MB
  unsigned short* Kb  = Qb + NE;                       //  8 MB
  unsigned short* Vb  = Kb + NE;                       //  8 MB
  unsigned short* Cb  = Vb + NE;                       //  8 MB
  unsigned short* Wqb = Cb + NE;                       // .5 MB
  unsigned short* Wkb = Wqb + WELEMS;                  // .5 MB
  unsigned short* Wvb = Wkb + WELEMS;                  // .5 MB
  unsigned short* Wob = Wvb + WELEMS;                  // .5 MB
  float*        Opre  = (float*)(Wob + WELEMS);        // 16 MB
  (void)in_sizes; (void)n_in; (void)out_size; (void)ws_size;

  // 0) one-shot bf16 materialization of x and weights
  k_cvt<<<dim3((unsigned)(NE / 2048)), 256, 0, stream>>>(x, xb);
  k_cvt<<<dim3(WELEMS / 2048), 256, 0, stream>>>(Wq, Wqb);
  k_cvt<<<dim3(WELEMS / 2048), 256, 0, stream>>>(Wk, Wkb);
  k_cvt<<<dim3(WELEMS / 2048), 256, 0, stream>>>(Wv, Wvb);
  k_cvt<<<dim3(WELEMS / 2048), 256, 0, stream>>>(Wo, Wob);
  // 1) Q/K/V projections (z selects matrix)
  k_qkv<<<dim3(DMODEL / 128, MROWS / 128, 3), 256, 0, stream>>>(
      xb, Wqb, Wkb, Wvb, bq, bk, bv, Qb, Kb, Vb);
  // 2) flash attention over 32 flat groups x 16 query tiles
  k_attn<<<dim3(SEQ / 128, NGROUP), 256, 0, stream>>>(Qb, Kb, Vb, Cb);
  // 3) output projection + bias + residual
  k_proj<<<dim3(DMODEL / 128, MROWS / 128), 256, 0, stream>>>(Cb, Wob, bo, x, Opre);
  // 4) LayerNorm
  k_ln<<<dim3(MROWS / 8), 256, 0, stream>>>(Opre, gamma, beta, out);
}
